// MultiHeadAttention_81973745811687
// MI455X (gfx1250) — matrix operand order repaired and checked by reading, not compiled, not run on hardware
//
#include <hip/hip_runtime.h>

// ---------------------------------------------------------------- types
typedef __attribute__((ext_vector_type(16))) __bf16 v16bf;
typedef __attribute__((ext_vector_type(8)))  float  v8f;

// ---------------------------------------------------------------- problem constants
constexpr int Bc    = 2;
constexpr int Sc    = 2048;
constexpr int Ec    = 2048;
constexpr int Hc    = 16;
constexpr int DHc   = 128;
constexpr int PASTc = 2048;
constexpr int KVc   = 4096;

// ---------------------------------------------------------------- helpers
// native bf16 convert (lowers to v_cvt_pk_bf16_f32 on gfx1250)
__device__ __forceinline__ unsigned short f2bf(float f) {
  union { __bf16 h; unsigned short u; } c;
  c.h = (__bf16)f;
  return c.u;
}

// A-fragment: two 8-half (16B) chunks (16-bit A 16x32 layout)
__device__ __forceinline__ v16bf frag_a(const unsigned short* p0,
                                        const unsigned short* p1) {
  union { v16bf v; uint4 q[2]; } u;
  u.q[0] = *(const uint4*)p0;
  u.q[1] = *(const uint4*)p1;
  return u.v;
}

// B-fragment: two 8-half (16B) chunks at p and p+16 (16-bit B 32x16 layout)
__device__ __forceinline__ v16bf frag_b(const unsigned short* p) {
  union { v16bf v; uint4 q[2]; } u;
  u.q[0] = *(const uint4*)p;
  u.q[1] = *(const uint4*)(p + 16);
  return u.v;
}

// gfx1250 async global->LDS copy, 16B per lane (ASYNCcnt-tracked).
// Takes the real shared-memory pointer: the ptrtoint captures the LDS object
// so the "memory" clobber is honored (keeps ds_load fragment reads alive),
// and the low 32 bits of the flat shared address ARE the LDS offset
// (ISA 10.2: LDS aperture maps addr[31:0] -> LDS).
__device__ __forceinline__ void async_ld16(const unsigned short* lds_ptr,
                                           const void* gptr) {
  unsigned off = (unsigned)(uintptr_t)lds_ptr;
  asm volatile("global_load_async_to_lds_b128 %0, %1, off"
               :: "v"(off), "v"(gptr) : "memory");
}
__device__ __forceinline__ void wait_async0() {
  asm volatile("s_wait_asynccnt 0x0" ::: "memory");
}

// ---------------------------------------------------------------- past-KV copy
__global__ __launch_bounds__(256) void copy_past_kernel(
    const float4* __restrict__ src, float4* __restrict__ dst) {
  size_t idx = (size_t)blockIdx.x * 256 + threadIdx.x;      // float4 units
  const size_t per_bh_src = (size_t)PASTc * DHc / 4;        // 65536
  const size_t per_bh_dst = (size_t)KVc * DHc / 4;          // 131072
  size_t bh = idx / per_bh_src;
  size_t r  = idx - bh * per_bh_src;
  dst[bh * per_bh_dst + r] = src[idx];
}

// ---------------------------------------------------------------- f32 -> bf16 bulk convert
__global__ __launch_bounds__(256) void convert_bf16_kernel(
    const float4* __restrict__ src, uint4* __restrict__ dst) {
  size_t i = (size_t)blockIdx.x * 256 + threadIdx.x;
  float4 a = src[2 * i];
  float4 b = src[2 * i + 1];
  union { unsigned short e[8]; uint4 q; } u;
  u.e[0] = f2bf(a.x); u.e[1] = f2bf(a.y); u.e[2] = f2bf(a.z); u.e[3] = f2bf(a.w);
  u.e[4] = f2bf(b.x); u.e[5] = f2bf(b.y); u.e[6] = f2bf(b.z); u.e[7] = f2bf(b.w);
  dst[i] = u.q;
}

// ---------------------------------------------------------------- GEMM: C = A @ W^T + bias
// Abf [M,K] bf16, Wbf [N,K] bf16. Async global->LDS staging, double-buffered,
// bf16 WMMA / f32 accumulate. MODE is compile-time:
//   MODE 0: Cf[m*N+n] (f32)
//   MODE 1: f32 scatter into KV cache [B,H,KV,DH] at row PAST+s
//   MODE 2: Cbf[m*N+n] = bf16((acc+bias)*oscale)   (pre-scaled Q)
template <int MODE>
__global__ __launch_bounds__(256) void gemm_bf16(
    const unsigned short* __restrict__ Abf,
    const unsigned short* __restrict__ Wbf,
    const float* __restrict__ bias,
    float* __restrict__ Cf, unsigned short* __restrict__ Cbf,
    int M, int N, int K, float oscale) {
  // layout: [buf(2)][mat(2)][row(128)][40 halves] (32 used, pad->40 = 80B rows)
  __shared__ __align__(16) unsigned short smem[2 * 2 * 128 * 40];

  const int tid  = threadIdx.x;
  const int lane = tid & 31;
  const int wave = tid >> 5;
  const int rowBlk = blockIdx.y * 128;   // M offset
  const int colBlk = blockIdx.x * 128;   // N offset
  const int wm = (wave >> 1) * 32;       // wave sub-tile: 32 rows x 64 cols
  const int wn = (wave & 1) * 64;

  v8f acc[2][4];
#pragma unroll
  for (int mi = 0; mi < 2; ++mi)
#pragma unroll
    for (int ni = 0; ni < 4; ++ni)
#pragma unroll
      for (int i = 0; i < 8; ++i) acc[mi][ni][i] = 0.0f;

  const int fr   = lane & 15;
  const int ka   = (lane >> 4) * 8;   // A chunk base (halves)
  const int kb16 = (lane >> 4) * 8;   // B chunk base (halves)

  // async staging: per matrix 512 x 16B chunks; thread owns chunks tid, tid+256
  auto stage_panel = [&](int buf, int kb) {
#pragma unroll
    for (int c = 0; c < 2; ++c) {
      int cid = c * 256 + tid;
      int row = cid >> 2;           // 0..127
      int seg = cid & 3;            // 16B segment within 64B row
      async_ld16(&smem[((buf * 2 + 0) * 128 + row) * 40 + seg * 8],
                 Abf + (size_t)(rowBlk + row) * K + kb + seg * 8);
      async_ld16(&smem[((buf * 2 + 1) * 128 + row) * 40 + seg * 8],
                 Wbf + (size_t)(colBlk + row) * K + kb + seg * 8);
    }
  };

  stage_panel(0, 0);
  wait_async0();
  __syncthreads();

  for (int kb = 0; kb < K; kb += 32) {
    const int cur  = (kb >> 5) & 1;
    const int nxt  = cur ^ 1;
    const bool more = (kb + 32) < K;

    if (more) stage_panel(nxt, kb + 32);   // async copies overlap WMMAs below

    const unsigned short* sa = &smem[(cur * 2 + 0) * 128 * 40];
    const unsigned short* sb = &smem[(cur * 2 + 1) * 128 * 40];

    v16bf afr[2], bfr[4];
#pragma unroll
    for (int mi = 0; mi < 2; ++mi) {
      const unsigned short* pr = sa + (wm + mi * 16 + fr) * 40;
      afr[mi] = frag_a(pr + ka, pr + ka + 16);
    }
#pragma unroll
    for (int ni = 0; ni < 4; ++ni)
      bfr[ni] = frag_b(sb + (wn + ni * 16 + fr) * 40 + kb16);

#pragma unroll
    for (int mi = 0; mi < 2; ++mi)
#pragma unroll
      for (int ni = 0; ni < 4; ++ni)
        acc[mi][ni] = __builtin_amdgcn_wmma_f32_16x16x32_bf16(
            false, afr[mi], false, bfr[ni], (short)0, acc[mi][ni], false, false);

    if (more) wait_async0();               // this wave's copies into nxt done
    __syncthreads();                       // publish to all waves
  }

  // epilogue: D layout: lane(0-15) N=lane, VGPR i -> M=i ; lanes 16-31 -> M=8+i
  const int n0    = colBlk + wn + fr;
  const int rbase = rowBlk + wm + ((lane >> 4) * 8);
#pragma unroll
  for (int mi = 0; mi < 2; ++mi)
#pragma unroll
    for (int ni = 0; ni < 4; ++ni) {
      int n = n0 + ni * 16;
      float bv = bias[n];
#pragma unroll
      for (int i = 0; i < 8; ++i) {
        int m = rbase + mi * 16 + i;
        float val = acc[mi][ni][i] + bv;
        if constexpr (MODE == 0) {
          Cf[(size_t)m * N + n] = val;
        } else if constexpr (MODE == 1) {
          int b = m >> 11, s = m & (Sc - 1);
          int h = n >> 7,  dh = n & (DHc - 1);
          Cf[(((size_t)(b * Hc + h) * KVc) + PASTc + s) * DHc + dh] = val;
        } else {
          Cbf[(size_t)m * N + n] = f2bf(val * oscale);
        }
      }
    }
}

// ---------------------------------------------------------------- flash attention
// Qbf: [B,S,E] bf16 (pre-scaled by 1/sqrt(DH)), K/V: [B,H,KV,DH] f32,
// Ctxbf out: [B,S,E] bf16
__global__ __launch_bounds__(256) void attn_flash(
    const unsigned short* __restrict__ Qbf, const float* __restrict__ Kc,
    const float* __restrict__ Vc, unsigned short* __restrict__ Ctxbf) {
  __shared__ __align__(16) unsigned short Ks[64][136];   // 128 used, pad->136
  __shared__ __align__(16) unsigned short Vt[128][72];   // 64 used, pad->72
  __shared__ __align__(16) unsigned short Ps[8][16][72];

  const int tid  = threadIdx.x;
  const int lane = tid & 31;
  const int wave = tid >> 5;
  const int bh   = blockIdx.y;
  const int b    = bh >> 4;
  const int h    = bh & (Hc - 1);
  const int qblk = blockIdx.x * 128;
  const int q0   = qblk + wave * 16;        // this wave's 16 query rows

  const int fr   = lane & 15;
  const int hsel = lane >> 4;

  // Q tile -> bf16 A-fragments straight from global (already scaled)
  const unsigned short* qrow =
      Qbf + ((size_t)(b * Sc + q0 + fr) * Ec) + h * DHc;
  v16bf qa[4];
#pragma unroll
  for (int c = 0; c < 4; ++c) {
    int k0 = c * 32 + hsel * 8;
    qa[c] = frag_a(qrow + k0, qrow + k0 + 16);
  }

  v8f o[8];
#pragma unroll
  for (int t = 0; t < 8; ++t)
#pragma unroll
    for (int i = 0; i < 8; ++i) o[t][i] = 0.0f;
  float mrow[8], lrow[8];
#pragma unroll
  for (int i = 0; i < 8; ++i) { mrow[i] = -3.0e38f; lrow[i] = 0.0f; }

  const float* Kbase = Kc + (size_t)bh * KVc * DHc;
  const float* Vbase = Vc + (size_t)bh * KVc * DHc;
  const int kvend = qblk + 127 + (KVc - Sc) + 1;   // causal exclusive bound

  const int srow = tid >> 2;            // staging: 0..63 kv row
  const int sdh0 = (tid & 3) * 32;      // 32 dh per thread

  for (int kv0 = 0; kv0 < kvend; kv0 += 64) {
    // ---- stage K (row-major) and V (transposed) into LDS as bf16
    const float* kp = Kbase + (size_t)(kv0 + srow) * DHc + sdh0;
    const float* vp = Vbase + (size_t)(kv0 + srow) * DHc + sdh0;
    if (kv0 + 64 < kvend) {             // WGP-scope prefetch of next chunk
      __builtin_prefetch(kp + 64 * DHc, 0, 3);
      __builtin_prefetch(vp + 64 * DHc, 0, 3);
    }
#pragma unroll
    for (int j = 0; j < 8; ++j) {
      float4 kf = ((const float4*)kp)[j];
      float4 vf = ((const float4*)vp)[j];
      int d = sdh0 + j * 4;
      Ks[srow][d + 0] = f2bf(kf.x); Ks[srow][d + 1] = f2bf(kf.y);
      Ks[srow][d + 2] = f2bf(kf.z); Ks[srow][d + 3] = f2bf(kf.w);
      Vt[d + 0][srow] = f2bf(vf.x); Vt[d + 1][srow] = f2bf(vf.y);
      Vt[d + 2][srow] = f2bf(vf.z); Vt[d + 3][srow] = f2bf(vf.w);
    }
    __syncthreads();

    // ---- scores: 16x64 tile = 4 N-tiles, K-dim = DH in 4 chunks of 32
    v8f st[4];
#pragma unroll
    for (int nt = 0; nt < 4; ++nt)
#pragma unroll
      for (int i = 0; i < 8; ++i) st[nt][i] = 0.0f;
#pragma unroll
    for (int c = 0; c < 4; ++c)
#pragma unroll
      for (int nt = 0; nt < 4; ++nt) {
        v16bf kb = frag_b(&Ks[nt * 16 + fr][c * 32 + hsel * 8]);
        st[nt] = __builtin_amdgcn_wmma_f32_16x16x32_bf16(
            false, qa[c], false, kb, (short)0, st[nt], false, false);
      }

    // ---- causal mask: masked where col > row + (KV - S)
    const int colb = kv0 + fr;
    const int rowb = q0 + hsel * 8 + (KVc - Sc);
#pragma unroll
    for (int nt = 0; nt < 4; ++nt) {
      int col = colb + nt * 16;
#pragma unroll
      for (int i = 0; i < 8; ++i)
        if (col > rowb + i) st[nt][i] = -3.0e38f;
    }

    // ---- online softmax (row spread across 16 lanes of a half-wave)
    float alpha[8];
#pragma unroll
    for (int i = 0; i < 8; ++i) {
      float v = fmaxf(fmaxf(st[0][i], st[1][i]), fmaxf(st[2][i], st[3][i]));
#pragma unroll
      for (int off = 1; off < 16; off <<= 1)
        v = fmaxf(v, __shfl_xor(v, off, 32));
      float mn = fmaxf(mrow[i], v);
      alpha[i] = __expf(mrow[i] - mn);
      mrow[i] = mn;
    }
#pragma unroll
    for (int nt = 0; nt < 4; ++nt)
#pragma unroll
      for (int i = 0; i < 8; ++i)
        st[nt][i] = __expf(st[nt][i] - mrow[i]);
#pragma unroll
    for (int i = 0; i < 8; ++i) {
      float s = st[0][i] + st[1][i] + st[2][i] + st[3][i];
#pragma unroll
      for (int off = 1; off < 16; off <<= 1)
        s += __shfl_xor(s, off, 32);
      lrow[i] = lrow[i] * alpha[i] + s;
    }
#pragma unroll
    for (int t = 0; t < 8; ++t)
#pragma unroll
      for (int i = 0; i < 8; ++i) o[t][i] *= alpha[i];

    // ---- P (D-layout) -> per-wave LDS pad -> A-fragments
#pragma unroll
    for (int nt = 0; nt < 4; ++nt) {
      int col = nt * 16 + fr;
#pragma unroll
      for (int i = 0; i < 8; ++i)
        Ps[wave][hsel * 8 + i][col] = f2bf(st[nt][i]);
    }
#pragma unroll
    for (int kc = 0; kc < 2; ++kc) {
      int k0 = kc * 32 + hsel * 8;
      v16bf pa = frag_a(&Ps[wave][fr][k0], &Ps[wave][fr][k0 + 16]);
#pragma unroll
      for (int dt = 0; dt < 8; ++dt) {
        v16bf vb = frag_b(&Vt[dt * 16 + fr][kc * 32 + hsel * 8]);
        o[dt] = __builtin_amdgcn_wmma_f32_16x16x32_bf16(
            false, pa, false, vb, (short)0, o[dt], false, false);
      }
    }
    __syncthreads();
  }

  // ---- normalize + write ctx [B,S,E] as bf16 (feeds O-projection)
  float inv[8];
#pragma unroll
  for (int i = 0; i < 8; ++i) inv[i] = 1.0f / lrow[i];
#pragma unroll
  for (int dt = 0; dt < 8; ++dt) {
    int dh = dt * 16 + fr;
#pragma unroll
    for (int i = 0; i < 8; ++i) {
      int row = q0 + hsel * 8 + i;
      Ctxbf[(size_t)(b * Sc + row) * Ec + h * DHc + dh] = f2bf(o[dt][i] * inv[i]);
    }
  }
}

// ---------------------------------------------------------------- launcher
extern "C" void kernel_launch(void* const* d_in, const int* in_sizes, int n_in,
                              void* d_out, int out_size, void* d_ws, size_t ws_size,
                              hipStream_t stream) {
  const float* x      = (const float*)d_in[0];
  const float* past_k = (const float*)d_in[1];
  const float* past_v = (const float*)d_in[2];
  const float* Wq = (const float*)d_in[3];  const float* bq = (const float*)d_in[4];
  const float* Wk = (const float*)d_in[5];  const float* bk = (const float*)d_in[6];
  const float* Wv = (const float*)d_in[7];  const float* bv = (const float*)d_in[8];
  const float* Wo = (const float*)d_in[9];  const float* bo = (const float*)d_in[10];

  float* out  = (float*)d_out;
  float* kout = out + (size_t)Bc * Sc * Ec;          // +8,388,608
  float* vout = kout + (size_t)Bc * Hc * KVc * DHc;  // +16,777,216

  const size_t nX = (size_t)Bc * Sc * Ec;            // 8,388,608
  const size_t nW = (size_t)Ec * Ec;                 // 4,194,304
  unsigned short* xbf   = (unsigned short*)d_ws;     // [B,S,E] bf16
  unsigned short* ctxbf = xbf;                       // alias: x dead after QKV gemms
  unsigned short* qbf   = xbf + nX;                  // [B,S,E] bf16, pre-scaled
  unsigned short* wbf   = qbf + nX;                  // [E,E] bf16, reused per weight

  const float qscale = 0.08838834764831845f;         // 1/sqrt(128)

  // 1) past KV -> caches
  copy_past_kernel<<<8192, 256, 0, stream>>>((const float4*)past_k, (float4*)kout);
  copy_past_kernel<<<8192, 256, 0, stream>>>((const float4*)past_v, (float4*)vout);

  // 2) activations + weights to bf16; QKV projections
  convert_bf16_kernel<<<(int)(nX / 2048), 256, 0, stream>>>(
      (const float4*)x, (uint4*)xbf);

  dim3 g(Ec / 128, (Bc * Sc) / 128);   // (16, 32)
  convert_bf16_kernel<<<(int)(nW / 2048), 256, 0, stream>>>(
      (const float4*)Wq, (uint4*)wbf);
  gemm_bf16<2><<<g, 256, 0, stream>>>(xbf, wbf, bq, nullptr, qbf,
                                      Bc * Sc, Ec, Ec, qscale);
  convert_bf16_kernel<<<(int)(nW / 2048), 256, 0, stream>>>(
      (const float4*)Wk, (uint4*)wbf);
  gemm_bf16<1><<<g, 256, 0, stream>>>(xbf, wbf, bk, kout, nullptr,
                                      Bc * Sc, Ec, Ec, 1.0f);
  convert_bf16_kernel<<<(int)(nW / 2048), 256, 0, stream>>>(
      (const float4*)Wv, (uint4*)wbf);
  gemm_bf16<1><<<g, 256, 0, stream>>>(xbf, wbf, bv, vout, nullptr,
                                      Bc * Sc, Ec, Ec, 1.0f);

  // 3) flash attention (writes bf16 ctx over xbf)
  dim3 ga(Sc / 128, Bc * Hc);          // (16, 32)
  attn_flash<<<ga, 256, 0, stream>>>(qbf, kout, vout, ctxbf);

  // 4) output projection
  convert_bf16_kernel<<<(int)(nW / 2048), 256, 0, stream>>>(
      (const float4*)Wo, (uint4*)wbf);
  gemm_bf16<0><<<g, 256, 0, stream>>>(ctxbf, wbf, bo, out, nullptr,
                                      Bc * Sc, Ec, Ec, 1.0f);
}
